// RecurrentTransformer_88450556494622
// MI455X (gfx1250) — matrix operand order repaired and checked by reading, not compiled, not run on hardware
//
#include <hip/hip_runtime.h>
#include <math.h>

// ===========================================================================
// RecurrentTransformer forward — MI455X (gfx1250) megakernel, v3.
//
// One workgroup (256 threads = 8 wave32) per batch element; per-sequence
// state lives in ~258KB LDS. All dense GEMMs (QKV / attention / out-proj /
// FF1 / FF2) run on v_wmma_f32_16x16x32_f16, f32 accumulate.
//
// v3: weights are converted ONCE per launch by a tiny prep kernel into d_ws
// as f16, column-major, K-padded — i.e. exactly the CDNA5 B-fragment layout.
// The megakernel then loads B fragments straight from global memory
// (2 x global_load_b128 per fragment, L2-resident: 192MB L2 vs ~380KB of
// weights), eliminating the per-block f32->f16 staging loops, the 51KB LDS
// B-buffer, and several barriers per layer. A operands stay in LDS
// (ds_load_b128). Requires ws_size >= ~380KB.
//
// Input pointer order assumption: d_in follows setup_inputs() dict insertion
// order, with the nested 'params' dict flattened in construction order:
//   0 loc  1 user  2 weekday  3 start_min  4 duration(unused)  5 diff  6 mask
//   7 loc_embed 8 user_embed 9 weekday_embed 10 time_embed 11 diff_embed
//   12 ctx_proj_w 13 ctx_proj_b 14 pos_embed 15 mem_init
//   16..27 layer0 {in_w,in_b,out_w,out_b,ln1_w,ln1_b,ln2_w,ln2_b,ff1_w,ff1_b,ff2_w,ff2_b}
//   28..39 layer1 (same)
//   40 norm_w 41 norm_b
//   42..45 mem_attn {in_w,in_b,out_w,out_b}
//   46 copy_q_w 47 copy_q_b 48 copy_k_w 49 copy_k_b
//   50 trans_embed 51 trans_fc_w 52 trans_fc_b 53 gate_w 54 gate_b
//   55 gen1_w 56 gen1_b 57 gen2_w 58 gen2_b 59 refine_w 60 refine_b
// ===========================================================================

#define EE    80      // embed dim
#define NHH   4       // heads
#define HDD   20      // head dim
#define HDP   32      // head dim padded to WMMA K
#define LL    100     // seq len
#define LP    112     // seq len padded to 16
#define FFD   320     // FF hidden
#define NLOCD 1200
#define EP    96      // E padded to 32 (WMMA K)
#define KP2   128     // key dim padded to 32 for P@V

// ---- LDS layout (byte offsets, all 16B aligned) ----
#define SM_X     0        // float[100*80]
#define SM_H     32000    // float[100*80]
#define SM_MEM   64000    // float[80]
#define SM_COMB  64320    // float[80]
#define SM_VA    64640    // float[80]
#define SM_VB    64960    // float[80]
#define SM_MSK   65280    // int[112]
#define SM_LOC   65728    // int[112]
#define SM_SC    66176    // float[256] misc scalars
#define SM_A     67200    // half[112*128]   A-operand staging / softmax P
#define SM_QP    95872    // half[4*112*32]  Q per-head padded      (28672 B)
#define SM_KP    124544   // half[4*112*32]  K per-head padded      (28672 B)
#define SM_VT    153216   // half[4*32*128]  V^T per-head padded    (32768 B)
// (FF hidden gH half[112*320]=71680B aliases SM_QP.. ; phases disjoint)
#define SM_AO    185984   // half[112*96]    attention output (f16 A operand)
#define SM_S     207488   // float[112*112]  scores (aliased by copy_logits)
#define SM_TOTAL 257664

// ---- d_ws layout (offsets in halfs): f16 column-major K-padded weights ----
#define WOFF_IN   0                       // per layer 240x96 = 23040
#define WSZ_IN    (240 * 96)
#define WOFF_OUT  (WOFF_IN + 2 * WSZ_IN)  // per layer 80x96 = 7680
#define WSZ_OUT   (80 * 96)
#define WOFF_FF1  (WOFF_OUT + 2 * WSZ_OUT)   // per layer 320x96 = 30720
#define WSZ_FF1   (320 * 96)
#define WOFF_FF2  (WOFF_FF1 + 2 * WSZ_FF1)   // per layer 80x320 = 25600
#define WSZ_FF2   (80 * 320)
#define WOFF_MEM  (WOFF_FF2 + 2 * WSZ_FF2)   // mem-attn K,V 160x96 = 15360
#define WSZ_MEM   (160 * 96)
#define WS_HALVES (WOFF_MEM + WSZ_MEM)       // 189440 halfs = 378880 bytes

typedef __attribute__((ext_vector_type(16))) _Float16 v16h;
typedef __attribute__((ext_vector_type(8)))  float    v8f;

struct RTParams {
  const int *loc, *user, *weekday, *start_min, *diff;
  const unsigned char *mask;
  const float *loc_embed, *user_embed, *weekday_embed, *time_embed, *diff_embed;
  const float *ctx_proj_w, *ctx_proj_b, *pos_embed, *mem_init;
  const float *in_b[2], *out_b[2];
  const float *ln1_w[2], *ln1_b[2], *ln2_w[2], *ln2_b[2];
  const float *ff1_b[2], *ff2_b[2];
  const float *norm_w, *norm_b;
  const float *mem_in_w, *mem_in_b, *mem_out_w, *mem_out_b;
  const float *copy_q_w, *copy_q_b, *copy_k_w, *copy_k_b;
  const float *trans_embed, *trans_fc_w, *trans_fc_b;
  const float *gate_w, *gate_b, *gen1_w, *gen1_b, *gen2_w, *gen2_b;
  const float *refine_w, *refine_b;
  // pre-converted f16 column-major weights in d_ws
  const _Float16 *w_in[2], *w_out[2], *w_ff1[2], *w_ff2[2], *w_memk, *w_memv;
  float *out;
};

__device__ __forceinline__ float gelu_f(float x) {
  return 0.5f * x * (1.0f + erff(x * 0.70710678118654752f));
}

// ---------------------------------------------------------------------------
// Prep: convert weight W (N rows x K cols, x@W.T convention) to f16 B-operand
// layout: dst[n*ldk + k] = W[n*Kk + k], zero-pad k in [Kk, ldk).
// ---------------------------------------------------------------------------
__global__ void rt_prep(const float *W, _Float16 *dst, int Nn, int Kk, int ldk) {
  const int i = blockIdx.x * 256 + threadIdx.x;
  if (i < Nn * ldk) {
    const int n = i / ldk, k = i % ldk;
    dst[i] = (k < Kk) ? (_Float16)W[n * Kk + k] : (_Float16)0.0f;
  }
}

// ---------------------------------------------------------------------------
// CDNA5 wave32 fragment layouts (CDNA5 ISA):
//   A (16x32 f16): lane l -> row M=l&15; elems e<8 -> K=k0+8*(l>>4)+e,
//                  elems e>=8 -> K=k0+16+8*(l>>4)+(e-8)     (two 8-half runs)
//   B (32x16 f16): lane l -> col N=l&15; same k order as A; code loads A and B as K=k0+16*(l>>4)+e
//                  => contiguous when B stored column-major (ldk halfs/col)
//   C/D (16x16 f32): vgpr r, lane l -> row r+8*(l>>4), col l&15
// A lives in LDS (ds_load_b128); B lives in global d_ws (global_load_b128).
// ---------------------------------------------------------------------------

// D = A(LDS,row-major,lda) * Bc(global,column-major,ldk) + bias -> f16 out.
__device__ void wmma_gemm_f16out(const _Float16 *A, int lda,
                                 const _Float16 *Bc, int ldk,
                                 const float *bias, _Float16 *C, int ldc,
                                 int Mt, int Nt, int Kt, int do_gelu, int tid) {
  const int wv = tid >> 5, lane = tid & 31, lh = lane >> 4, lm = lane & 15;
  for (int t = wv; t < Mt * Nt; t += 8) {
    const int mi = t / Nt, ni = t % Nt;
    v8f acc = {0.f, 0.f, 0.f, 0.f, 0.f, 0.f, 0.f, 0.f};
    for (int kt = 0; kt < Kt; ++kt) {
      const int k0 = kt * 32;
      v16h af, bf;
      const _Float16 *ap = A + (mi * 16 + lm) * lda + k0 + 16 * lh;
#pragma unroll
      for (int e = 0; e < 16; ++e) af[e] = ap[e];
      const _Float16 *bp = Bc + (ni * 16 + lm) * ldk + k0 + 16 * lh;
#pragma unroll
      for (int e = 0; e < 16; ++e) bf[e] = bp[e];
      acc = __builtin_amdgcn_wmma_f32_16x16x32_f16(false, af, false, bf,
                                                   (short)0, acc, false, false);
    }
#pragma unroll
    for (int r = 0; r < 8; ++r) {
      const int row = mi * 16 + 8 * lh + r, col = ni * 16 + lm;
      float v = acc[r] + (bias ? bias[col] : 0.0f);
      if (do_gelu) v = gelu_f(v);
      C[row * ldc + col] = (_Float16)v;
    }
  }
}

// C(f32,row-major) += A * Bc + bias, only rows < rowlim stored.
__device__ void wmma_gemm_accf32(const _Float16 *A, int lda,
                                 const _Float16 *Bc, int ldk,
                                 const float *bias, float *C, int ldc,
                                 int Mt, int Nt, int Kt, int rowlim, int tid) {
  const int wv = tid >> 5, lane = tid & 31, lh = lane >> 4, lm = lane & 15;
  for (int t = wv; t < Mt * Nt; t += 8) {
    const int mi = t / Nt, ni = t % Nt;
    v8f acc = {0.f, 0.f, 0.f, 0.f, 0.f, 0.f, 0.f, 0.f};
    for (int kt = 0; kt < Kt; ++kt) {
      const int k0 = kt * 32;
      v16h af, bf;
      const _Float16 *ap = A + (mi * 16 + lm) * lda + k0 + 16 * lh;
#pragma unroll
      for (int e = 0; e < 16; ++e) af[e] = ap[e];
      const _Float16 *bp = Bc + (ni * 16 + lm) * ldk + k0 + 16 * lh;
#pragma unroll
      for (int e = 0; e < 16; ++e) bf[e] = bp[e];
      acc = __builtin_amdgcn_wmma_f32_16x16x32_f16(false, af, false, bf,
                                                   (short)0, acc, false, false);
    }
#pragma unroll
    for (int r = 0; r < 8; ++r) {
      const int row = mi * 16 + 8 * lh + r, col = ni * 16 + lm;
      if (row < rowlim) C[row * ldc + col] += acc[r] + bias[col];
    }
  }
}

// Projection GEMM (M=112,N=80,K=96) with head-split store.
// mode 0: out[hh*LP*HDP + row*HDP + d]   (Q,K per-head padded, row-major)
// mode 1: out[hh*HDP*KP2 + d*KP2 + row]  (V transposed per-head padded)
__device__ void wmma_gemm_qkv(const _Float16 *A, int lda,
                              const _Float16 *Bc, int ldk,
                              const float *bias, _Float16 *out, int mode,
                              int tid) {
  const int wv = tid >> 5, lane = tid & 31, lh = lane >> 4, lm = lane & 15;
  for (int t = wv; t < 7 * 5; t += 8) {
    const int mi = t / 5, ni = t % 5;
    v8f acc = {0.f, 0.f, 0.f, 0.f, 0.f, 0.f, 0.f, 0.f};
    for (int kt = 0; kt < 3; ++kt) {
      const int k0 = kt * 32;
      v16h af, bf;
      const _Float16 *ap = A + (mi * 16 + lm) * lda + k0 + 16 * lh;
#pragma unroll
      for (int e = 0; e < 16; ++e) af[e] = ap[e];
      const _Float16 *bp = Bc + (ni * 16 + lm) * ldk + k0 + 16 * lh;
#pragma unroll
      for (int e = 0; e < 16; ++e) bf[e] = bp[e];
      acc = __builtin_amdgcn_wmma_f32_16x16x32_f16(false, af, false, bf,
                                                   (short)0, acc, false, false);
    }
    const int col = ni * 16 + lm;   // 0..79
    const int hh = col / HDD, d = col % HDD;
#pragma unroll
    for (int r = 0; r < 8; ++r) {
      const int row = mi * 16 + 8 * lh + r;
      const _Float16 v = (_Float16)(acc[r] + bias[col]);
      if (mode == 0) out[hh * LP * HDP + row * HDP + d] = v;
      else           out[hh * HDP * KP2 + d * KP2 + row] = v;
    }
  }
}

// S[112x112] = (Q_h K_h^T) * scale, all operands zero-padded; no guards.
__device__ void attn_scores(const _Float16 *Qp, const _Float16 *Kp, float *S,
                            int hh, float scale, int tid) {
  const int wv = tid >> 5, lane = tid & 31, lh = lane >> 4, lm = lane & 15;
  const _Float16 *Qh = Qp + hh * LP * HDP;
  const _Float16 *Kh = Kp + hh * LP * HDP;
  for (int t = wv; t < 49; t += 8) {
    const int mi = t / 7, ni = t % 7;
    v8f acc = {0.f, 0.f, 0.f, 0.f, 0.f, 0.f, 0.f, 0.f};
    v16h af, bf;
    const _Float16 *ap = Qh + (mi * 16 + lm) * HDP + 16 * lh;
#pragma unroll
    for (int e = 0; e < 16; ++e) af[e] = ap[e];
    const _Float16 *bp = Kh + (ni * 16 + lm) * HDP + 16 * lh;
#pragma unroll
    for (int e = 0; e < 16; ++e) bf[e] = bp[e];
    acc = __builtin_amdgcn_wmma_f32_16x16x32_f16(false, af, false, bf,
                                                 (short)0, acc, false, false);
#pragma unroll
    for (int r = 0; r < 8; ++r)
      S[(mi * 16 + 8 * lh + r) * LP + ni * 16 + lm] = acc[r] * scale;
  }
}

// AO[:, hh*20..hh*20+19] = P[112x128] @ V_h ; V^T layout, zero-padded.
__device__ void attn_av(const _Float16 *Pm, const _Float16 *Vt, _Float16 *AO,
                        int hh, int tid) {
  const int wv = tid >> 5, lane = tid & 31, lh = lane >> 4, lm = lane & 15;
  const _Float16 *Vh = Vt + hh * HDP * KP2;
  for (int t = wv; t < 14; t += 8) {
    const int mi = t / 2, ni = t % 2;
    v8f acc = {0.f, 0.f, 0.f, 0.f, 0.f, 0.f, 0.f, 0.f};
    for (int kt = 0; kt < 4; ++kt) {
      const int k0 = kt * 32;
      v16h af, bf;
      const _Float16 *ap = Pm + (mi * 16 + lm) * KP2 + k0 + 16 * lh;
#pragma unroll
      for (int e = 0; e < 16; ++e) af[e] = ap[e];
      const _Float16 *bp = Vh + (ni * 16 + lm) * KP2 + k0 + 16 * lh;
#pragma unroll
      for (int e = 0; e < 16; ++e) bf[e] = bp[e];
      acc = __builtin_amdgcn_wmma_f32_16x16x32_f16(false, af, false, bf,
                                                   (short)0, acc, false, false);
    }
#pragma unroll
    for (int r = 0; r < 8; ++r) {
      const int row = mi * 16 + 8 * lh + r, col = ni * 16 + lm;
      if (col < HDD) AO[row * EP + hh * HDD + col] = (_Float16)acc[r];
    }
  }
}

// LayerNorm rows of X[100x80] -> f16 A-staging [112x96] with zero padding.
__device__ void ln_stage(const float *X, const float *w, const float *b,
                         _Float16 *Ah, int tid) {
  if (tid < LP) {
    const int r = tid;
    if (r < LL) {
      const float *xr = X + r * EE;
      float m = 0.f;
      for (int e = 0; e < EE; ++e) m += xr[e];
      m *= (1.0f / EE);
      float v = 0.f;
      for (int e = 0; e < EE; ++e) { const float d = xr[e] - m; v += d * d; }
      v *= (1.0f / EE);
      const float inv = rsqrtf(v + 1e-5f);
      for (int e = 0; e < EE; ++e)
        Ah[r * EP + e] = (_Float16)((xr[e] - m) * inv * w[e] + b[e]);
      for (int e = EE; e < EP; ++e) Ah[r * EP + e] = (_Float16)0.0f;
    } else {
      for (int e = 0; e < EP; ++e) Ah[r * EP + e] = (_Float16)0.0f;
    }
  }
}

__device__ void zero_lds32(int *p, int n32, int tid) {
  for (int i = tid; i < n32; i += 256) p[i] = 0;
}

__global__ __launch_bounds__(256, 1)
void rt_forward(RTParams p) {
  const int b = blockIdx.x, tid = threadIdx.x;
  extern __shared__ unsigned char smem[];
  float    *s_x    = (float *)(smem + SM_X);
  float    *s_h    = (float *)(smem + SM_H);
  float    *s_mem  = (float *)(smem + SM_MEM);
  float    *s_comb = (float *)(smem + SM_COMB);
  float    *s_vA   = (float *)(smem + SM_VA);
  float    *s_vB   = (float *)(smem + SM_VB);
  int      *s_msk  = (int *)(smem + SM_MSK);
  int      *s_loc  = (int *)(smem + SM_LOC);
  float    *s_sc   = (float *)(smem + SM_SC);
  _Float16 *s_a    = (_Float16 *)(smem + SM_A);
  _Float16 *s_qp   = (_Float16 *)(smem + SM_QP);
  _Float16 *s_kp   = (_Float16 *)(smem + SM_KP);
  _Float16 *s_vt   = (_Float16 *)(smem + SM_VT);
  _Float16 *s_g    = (_Float16 *)(smem + SM_QP);   // FF hidden aliases Q/K/V
  _Float16 *s_ao   = (_Float16 *)(smem + SM_AO);
  float    *s_S    = (float *)(smem + SM_S);
  float    *s_cl   = s_S;                           // copy_logits alias

  // Warm L2 with the f16 weight blob (global_prefetch_b8 path).
  for (int off = tid * 128; off < 2 * WSZ_IN * 2; off += 256 * 128)
    __builtin_prefetch((const unsigned char *)p.w_in[0] + off, 0, 1);

  // ---- Phase 0: masks/locs + x = loc_embed + ctx@Wc^T + bc + pos ----
  if (tid < LP) {
    s_msk[tid] = (tid < LL) ? (p.mask[b * LL + tid] ? 1 : 0) : 0;
    s_loc[tid] = (tid < LL) ? p.loc[b * LL + tid] : 0;
  }
  __syncthreads();
  for (int i = tid; i < LL * EE; i += 256) {
    const int j = i / EE, e = i % EE;
    const int u = p.user[b * LL + j];
    const int wd = p.weekday[b * LL + j];
    int tm = p.start_min[b * LL + j] / 30; tm = tm < 0 ? 0 : (tm > 47 ? 47 : tm);
    int df = p.diff[b * LL + j];           df = df < 0 ? 0 : (df > 49 ? 49 : df);
    const float *cw = p.ctx_proj_w + e * 56;
    float acc = p.ctx_proj_b[e];
    const float *ue = p.user_embed + u * 24;
    for (int c = 0; c < 24; ++c) acc += ue[c] * cw[c];
    const float *we = p.weekday_embed + wd * 8;
    for (int c = 0; c < 8; ++c) acc += we[c] * cw[24 + c];
    const float *te = p.time_embed + tm * 16;
    for (int c = 0; c < 16; ++c) acc += te[c] * cw[32 + c];
    const float *de = p.diff_embed + df * 8;
    for (int c = 0; c < 8; ++c) acc += de[c] * cw[48 + c];
    s_x[i] = p.loc_embed[s_loc[j] * EE + e] + acc + p.pos_embed[j * EE + e];
  }
  if (tid < EE) s_mem[tid] = p.mem_init[tid];
  __syncthreads();

  const float scl20 = 0.22360679774997896f;  // 1/sqrt(20)

  // ---- recurrent cycles ----
  for (int cyc = 0; cyc < 2; ++cyc) {
    for (int i = tid; i < LL * EE; i += 256)
      s_h[i] = s_x[i] + 0.2f * s_mem[i % EE];
    __syncthreads();

    for (int ly = 0; ly < 2; ++ly) {
      // LN1 -> f16 A ; zero Q/K/V region (pads must be clean zeros)
      ln_stage(s_h, p.ln1_w[ly], p.ln1_b[ly], s_a, tid);
      zero_lds32((int *)s_qp, (SM_AO - SM_QP) / 4, tid);
      __syncthreads();
      // Q, K, V projections: disjoint outputs, same A -> no barriers between
      wmma_gemm_qkv(s_a, EP, p.w_in[ly] + 0 * EE * EP, EP, p.in_b[ly] + 0 * EE,
                    s_qp, 0, tid);
      wmma_gemm_qkv(s_a, EP, p.w_in[ly] + 1 * EE * EP, EP, p.in_b[ly] + 1 * EE,
                    s_kp, 0, tid);
      wmma_gemm_qkv(s_a, EP, p.w_in[ly] + 2 * EE * EP, EP, p.in_b[ly] + 2 * EE,
                    s_vt, 1, tid);
      __syncthreads();
      // zero attention-output staging (incl. pad cols 80..95)
      zero_lds32((int *)s_ao, LP * EP / 2, tid);
      __syncthreads();

      for (int hh = 0; hh < NHH; ++hh) {
        attn_scores(s_qp, s_kp, s_S, hh, scl20, tid);
        __syncthreads();
        // masked softmax rows -> P (f16, [112x128], zero-padded)
        if (tid < LP) {
          const int r = tid;
          if (r < LL) {
            float mx = -3.0e38f;
            for (int j = 0; j < LL; ++j) {
              const float v = s_msk[j] ? s_S[r * LP + j] : -1e9f;
              if (v > mx) mx = v;
            }
            float sm = 0.f;
            for (int j = 0; j < LL; ++j) {
              const float v = s_msk[j] ? s_S[r * LP + j] : -1e9f;
              sm += expf(v - mx);
            }
            const float inv = 1.0f / sm;
            for (int j = 0; j < LL; ++j) {
              const float v = s_msk[j] ? s_S[r * LP + j] : -1e9f;
              s_a[r * KP2 + j] = (_Float16)(expf(v - mx) * inv);
            }
            for (int j = LL; j < KP2; ++j) s_a[r * KP2 + j] = (_Float16)0.0f;
          } else {
            for (int j = 0; j < KP2; ++j) s_a[r * KP2 + j] = (_Float16)0.0f;
          }
        }
        __syncthreads();
        attn_av(s_a, s_vt, s_ao, hh, tid);
        __syncthreads();
      }
      // out-proj + residual into h (WMMA, f32 accumulate)
      wmma_gemm_accf32(s_ao, EP, p.w_out[ly], EP, p.out_b[ly], s_h, EE,
                       7, 5, 3, LL, tid);
      __syncthreads();

      // LN2 -> f16 A
      ln_stage(s_h, p.ln2_w[ly], p.ln2_b[ly], s_a, tid);
      __syncthreads();
      // FF1 (fused bias + exact GELU, f16 out), 2 N-chunks, no barrier between
      wmma_gemm_f16out(s_a, EP, p.w_ff1[ly] + 0 * 160 * EP, EP,
                       p.ff1_b[ly] + 0, s_g + 0, FFD, 7, 10, 3, 1, tid);
      wmma_gemm_f16out(s_a, EP, p.w_ff1[ly] + 160 * EP, EP,
                       p.ff1_b[ly] + 160, s_g + 160, FFD, 7, 10, 3, 1, tid);
      __syncthreads();
      // FF2 + residual into h (B column-major, ldk = 320)
      wmma_gemm_accf32(s_g, FFD, p.w_ff2[ly], FFD, p.ff2_b[ly], s_h, EE,
                       7, 5, 10, LL, tid);
      __syncthreads();
    }

    // final LN in place on h
    if (tid < LL) {
      float *xr = s_h + tid * EE;
      float m = 0.f;
      for (int e = 0; e < EE; ++e) m += xr[e];
      m *= (1.0f / EE);
      float v = 0.f;
      for (int e = 0; e < EE; ++e) { const float d = xr[e] - m; v += d * d; }
      v *= (1.0f / EE);
      const float inv = rsqrtf(v + 1e-5f);
      for (int e = 0; e < EE; ++e)
        xr[e] = (xr[e] - m) * inv * p.norm_w[e] + p.norm_b[e];
    }
    __syncthreads();

    // ---- memory attention: mem (len-1 query) over h ----
    if (tid < LP) {  // stage h as f16 A
      const int r = tid;
      if (r < LL) {
        for (int e = 0; e < EE; ++e) s_a[r * EP + e] = (_Float16)s_h[r * EE + e];
        for (int e = EE; e < EP; ++e) s_a[r * EP + e] = (_Float16)0.0f;
      } else {
        for (int e = 0; e < EP; ++e) s_a[r * EP + e] = (_Float16)0.0f;
      }
    }
    zero_lds32((int *)s_kp, (SM_AO - SM_KP) / 4, tid);  // K + V^T regions
    __syncthreads();
    wmma_gemm_qkv(s_a, EP, p.w_memk, EP, p.mem_in_b + EE, s_kp, 0, tid);
    wmma_gemm_qkv(s_a, EP, p.w_memv, EP, p.mem_in_b + 2 * EE, s_vt, 1, tid);
    __syncthreads();
    if (tid < EE) {  // q = mem@Wq^T + bq
      float a = p.mem_in_b[tid];
      for (int k = 0; k < EE; ++k) a += s_mem[k] * p.mem_in_w[tid * EE + k];
      s_vA[tid] = a;
    }
    __syncthreads();
    if (tid < NHH * LL) {  // masked scores per head
      const int hh = tid / LL, j = tid % LL;
      float a = 0.f;
      for (int d = 0; d < HDD; ++d)
        a += s_vA[hh * HDD + d] * (float)s_kp[hh * LP * HDP + j * HDP + d];
      a *= scl20;
      if (!s_msk[j]) a = -1e9f;
      s_S[hh * LP + j] = a;
    }
    __syncthreads();
    if (tid < NHH) {  // per-head softmax
      const int hh = tid;
      float mx = -3.0e38f;
      for (int j = 0; j < LL; ++j) if (s_S[hh * LP + j] > mx) mx = s_S[hh * LP + j];
      float sm = 0.f;
      for (int j = 0; j < LL; ++j) sm += expf(s_S[hh * LP + j] - mx);
      const float inv = 1.0f / sm;
      for (int j = 0; j < LL; ++j) s_S[hh * LP + j] = expf(s_S[hh * LP + j] - mx) * inv;
    }
    __syncthreads();
    if (tid < EE) {  // o = A @ V_h  (V^T layout: [hh][d][j])
      const int hh = tid / HDD, d = tid % HDD;
      float a = 0.f;
      for (int j = 0; j < LL; ++j)
        a += s_S[hh * LP + j] * (float)s_vt[hh * HDP * KP2 + d * KP2 + j];
      s_vB[tid] = a;
    }
    __syncthreads();
    if (tid < EE) {  // mem += o@Wo^T + bo [+ mem_init[cyc+1]]
      float a = p.mem_out_b[tid];
      for (int k = 0; k < EE; ++k) a += s_vB[k] * p.mem_out_w[tid * EE + k];
      float nm = s_mem[tid] + a;
      if (cyc + 1 < 2) nm += p.mem_init[(cyc + 1) * EE + tid];
      s_mem[tid] = nm;
    }
    __syncthreads();
  }

  // ---- tail: combined, refine, copy, generator, gate ----
  if (tid == 0) {
    int n = 0;
    for (int j = 0; j < LL; ++j) n += s_msk[j];
    int last = n - 1; if (last < 0) last = 0;
    s_loc[110] = last;
  }
  __syncthreads();
  const int last = s_loc[110];
  if (tid < EE) s_comb[tid] = s_h[last * EE + tid] + s_mem[tid];
  __syncthreads();
  for (int it = 0; it < 12; ++it) {
    if (tid < EE) {
      float a = p.refine_b[tid];
      for (int k = 0; k < EE; ++k) a += s_comb[k] * p.refine_w[tid * EE + k];
      s_vA[tid] = gelu_f(a);
    }
    __syncthreads();
    if (tid < EE) s_comb[tid] += 0.1f * s_vA[tid];
    __syncthreads();
  }
  if (tid < EE) {  // query = comb@copy_q^T + b
    float a = p.copy_q_b[tid];
    for (int k = 0; k < EE; ++k) a += s_comb[k] * p.copy_q_w[tid * EE + k];
    s_vA[tid] = a;
  }
  __syncthreads();
  if (tid < EE) {  // qk[k] = sum_e q[e]*ck_w[e][k]
    float a = 0.f;
    for (int e = 0; e < EE; ++e) a += s_vA[e] * p.copy_k_w[e * EE + tid];
    s_vB[tid] = a;
  }
  if (tid == 0) {  // qb = q . ck_b
    float a = 0.f;
    for (int e = 0; e < EE; ++e) a += s_vA[e] * p.copy_k_b[e];
    s_sc[0] = a;
  }
  __syncthreads();
  for (int i = tid; i < NLOCD; i += 256) s_cl[i] = 0.0f;
  __syncthreads();
  const float iscl = 0.11180339887498949f;  // 1/sqrt(80)
  if (tid < LL && s_msk[tid]) {
    float a = s_sc[0];
    for (int k = 0; k < EE; ++k) a += s_vB[k] * s_h[tid * EE + k];
    a *= iscl;
    // scatter-max; safe as signed-int max since logits are clamped at 0
    atomicMax((int *)&s_cl[s_loc[tid]], __float_as_int(a));
  }
  __syncthreads();
  if (tid < EE) {  // trans_feat
    const int tl = s_loc[last];
    float a = p.trans_fc_b[tid];
    for (int k = 0; k < 40; ++k)
      a += p.trans_embed[tl * 40 + k] * p.trans_fc_w[tid * 40 + k];
    s_vA[tid] = a;
  }
  __syncthreads();
  if (tid < EE) {  // gen_h
    float a = p.gen1_b[tid];
    for (int k = 0; k < EE; ++k) a += (s_comb[k] + s_vA[k]) * p.gen1_w[tid * EE + k];
    s_vB[tid] = gelu_f(a);
  }
  if (tid == 0) {  // gate
    float a = p.gate_b[0];
    for (int k = 0; k < EE; ++k)
      a += s_comb[k] * p.gate_w[k] + s_vA[k] * p.gate_w[EE + k];
    s_sc[1] = 1.0f / (1.0f + expf(-a));
  }
  __syncthreads();
  const float gate = s_sc[1];
  for (int l = tid; l < NLOCD; l += 256) {
    float a = p.gen2_b[l];
    const float *w = p.gen2_w + l * EE;
    for (int k = 0; k < EE; ++k) a += s_vB[k] * w[k];
    p.out[(size_t)b * NLOCD + l] = gate * s_cl[l] + (1.0f - gate) * a;
  }
}

extern "C" void kernel_launch(void *const *d_in, const int *in_sizes, int n_in,
                              void *d_out, int out_size, void *d_ws,
                              size_t ws_size, hipStream_t stream) {
  (void)n_in; (void)out_size; (void)ws_size;  // need ws_size >= ~380KB
  RTParams p;
  p.loc       = (const int *)d_in[0];
  p.user      = (const int *)d_in[1];
  p.weekday   = (const int *)d_in[2];
  p.start_min = (const int *)d_in[3];
  // d_in[4] = duration (unused by the reference forward)
  p.diff      = (const int *)d_in[5];
  p.mask      = (const unsigned char *)d_in[6];
  const float *in_w[2], *out_w[2], *ff1_w[2], *ff2_w[2];
  int i = 7;
  p.loc_embed     = (const float *)d_in[i++];
  p.user_embed    = (const float *)d_in[i++];
  p.weekday_embed = (const float *)d_in[i++];
  p.time_embed    = (const float *)d_in[i++];
  p.diff_embed    = (const float *)d_in[i++];
  p.ctx_proj_w    = (const float *)d_in[i++];
  p.ctx_proj_b    = (const float *)d_in[i++];
  p.pos_embed     = (const float *)d_in[i++];
  p.mem_init      = (const float *)d_in[i++];
  for (int ly = 0; ly < 2; ++ly) {
    in_w[ly]    = (const float *)d_in[i++];
    p.in_b[ly]  = (const float *)d_in[i++];
    out_w[ly]   = (const float *)d_in[i++];
    p.out_b[ly] = (const float *)d_in[i++];
    p.ln1_w[ly] = (const float *)d_in[i++];
    p.ln1_b[ly] = (const float *)d_in[i++];
    p.ln2_w[ly] = (const float *)d_in[i++];
    p.ln2_b[ly] = (const float *)d_in[i++];
    ff1_w[ly]   = (const float *)d_in[i++];
    p.ff1_b[ly] = (const float *)d_in[i++];
    ff2_w[ly]   = (const float *)d_in[i++];
    p.ff2_b[ly] = (const float *)d_in[i++];
  }
  p.norm_w      = (const float *)d_in[i++];
  p.norm_b      = (const float *)d_in[i++];
  p.mem_in_w    = (const float *)d_in[i++];
  p.mem_in_b    = (const float *)d_in[i++];
  p.mem_out_w   = (const float *)d_in[i++];
  p.mem_out_b   = (const float *)d_in[i++];
  p.copy_q_w    = (const float *)d_in[i++];
  p.copy_q_b    = (const float *)d_in[i++];
  p.copy_k_w    = (const float *)d_in[i++];
  p.copy_k_b    = (const float *)d_in[i++];
  p.trans_embed = (const float *)d_in[i++];
  p.trans_fc_w  = (const float *)d_in[i++];
  p.trans_fc_b  = (const float *)d_in[i++];
  p.gate_w      = (const float *)d_in[i++];
  p.gate_b      = (const float *)d_in[i++];
  p.gen1_w      = (const float *)d_in[i++];
  p.gen1_b      = (const float *)d_in[i++];
  p.gen2_w      = (const float *)d_in[i++];
  p.gen2_b      = (const float *)d_in[i++];
  p.refine_w    = (const float *)d_in[i++];
  p.refine_b    = (const float *)d_in[i++];
  p.out = (float *)d_out;

  // ---- prep pass: weights -> f16 column-major K-padded in d_ws ----
  _Float16 *ws = (_Float16 *)d_ws;
  auto conv = [&](const float *W, _Float16 *dst, int Nn, int Kk, int ldk) {
    const int n = Nn * ldk;
    rt_prep<<<(n + 255) / 256, 256, 0, stream>>>(W, dst, Nn, Kk, ldk);
  };
  for (int ly = 0; ly < 2; ++ly) {
    p.w_in[ly]  = ws + WOFF_IN + ly * WSZ_IN;
    p.w_out[ly] = ws + WOFF_OUT + ly * WSZ_OUT;
    p.w_ff1[ly] = ws + WOFF_FF1 + ly * WSZ_FF1;
    p.w_ff2[ly] = ws + WOFF_FF2 + ly * WSZ_FF2;
    conv(in_w[ly],  (_Float16 *)p.w_in[ly],  240, 80, 96);
    conv(out_w[ly], (_Float16 *)p.w_out[ly],  80, 80, 96);
    conv(ff1_w[ly], (_Float16 *)p.w_ff1[ly], 320, 80, 96);
    conv(ff2_w[ly], (_Float16 *)p.w_ff2[ly],  80, 320, 320);
  }
  p.w_memk = ws + WOFF_MEM;
  p.w_memv = ws + WOFF_MEM + 80 * 96;
  conv(p.mem_in_w + 80 * 80, (_Float16 *)p.w_memk, 160, 80, 96);

  const int B = in_sizes[0] / LL;
  (void)hipFuncSetAttribute((const void *)rt_forward,
                            hipFuncAttributeMaxDynamicSharedMemorySize,
                            SM_TOTAL);
  rt_forward<<<B, 256, SM_TOTAL, stream>>>(p);
}
